// RLN_11364483465499
// MI455X (gfx1250) — matrix operand order repaired and checked by reading, not compiled, not run on hardware
//
#include <hip/hip_runtime.h>

// ---------------- CDNA5 WMMA types / helpers ----------------
typedef __attribute__((ext_vector_type(16))) __bf16 v16bf;
typedef __attribute__((ext_vector_type(8)))  __bf16 v8bf;
typedef __attribute__((ext_vector_type(4)))  __bf16 v4bf;
typedef __attribute__((ext_vector_type(8)))  float  v8f;

__device__ __forceinline__ __bf16 f2bf(float f) { return (__bf16)f; }  // native v_cvt

__device__ __forceinline__ v16bf combine16(v8bf lo, v8bf hi) {
  return __builtin_shufflevector(lo, hi, 0, 1, 2, 3, 4, 5, 6, 7,
                                         8, 9, 10, 11, 12, 13, 14, 15);
}
// fast transcendentals: v_exp_f32 + v_rcp_f32, no libm branches / IEEE division
__device__ __forceinline__ float rcpf_(float x) { return __builtin_amdgcn_rcpf(x); }
__device__ __forceinline__ float sigmf(float x) { return rcpf_(1.0f + __expf(-x)); }
__device__ __forceinline__ float tanh_fast(float x) {
  float e = __expf(2.0f * x);
  return 1.0f - 2.0f * rcpf_(e + 1.0f);
}
__device__ __forceinline__ float leakyf(float v) { return v >= 0.0f ? v : 0.01f * v; }

#define WMMA_BF16(A, Bm, Cm) \
  __builtin_amdgcn_wmma_f32_16x16x32_bf16(false, (A), false, (Bm), (short)0, (Cm), false, false)

// ---------------- problem constants ----------------
#define BSZ   2048
#define CCH   124
#define FIN   32
#define TSEQ  248
#define HL    64
#define G4    256
#define DMLP  600
#define NCLS  6

// ---------------- LDS layout (bytes) ----------------
// B-weight buffers hold pre-swizzled WMMA B fragments: tile tt -> 512 bf16,
// lane l owns 16 contiguous elements at tt*512 + l*16 (one 32B run -> 2x b128).
#define SEQ_OFF    0         // bf16 [16][248][16]            = 126976
#define WHH_OFF    126976    // bf16 swz 32 tiles (2kc x 16)  = 32768
#define WIH_OFF    159744    // bf16 swz 16 tiles (K pad 32)  = 16384
#define WV_OFF     176128    // bf16 swz 2 tiles              = 2048
#define WO_OFF     178176    // bf16 swz 2 tiles              = 2048
#define WP_OFF     180224    // bf16 swz 1 tile               = 1024
#define FFW_OFF    181248    // bf16 swz 16 tiles             = 16384
#define BIAS_OFF   197632    // f32 [1200]                    = 4800
#define GATE_OFF   202432    // f32 col-major [256][16]       = 16384
#define HBUF_OFF   218816    // bf16 [16][64]                 = 2048
#define SCR_OFF    220864    // bf16 8 waves x 1024           = 16384
#define M1_OFF     237248    // f32 col-major [608][16]       = 38912
#define LDS_BYTES  276160
// MLP overlays (seq region dead after LSTM):
#define W1_OFF     0         // bf16 swz 76 tiles (2kc x 38)  = 77824
#define W2_OFF     77824     // f32 [600][6]                  = 14400

// bias sub-offsets (floats)
#define BV_B   0
#define BO_B   32
#define BP_B   64
#define FFB_B  80
#define LB_B   336
#define B1_B   592
#define B2_B   1192

// Swizzle a row-major f32 weight [krows x ncols] (ld = ncols) into WMMA-B
// fragment order, zero-padding K up to kchunks*32 and N up to ntiles*16.
__device__ __forceinline__ void swizzleB(__bf16* dst, const float* W, int ld,
                                         int krows, int ncols, int ntiles,
                                         int kchunks, int tid) {
  int total = kchunks * ntiles * 512;
  for (int d = tid; d < total; d += 256) {
    int tt = d >> 9;
    int l  = (d >> 4) & 31;
    int e  = d & 15;
    int kc = tt / ntiles, nt = tt - kc * ntiles;
    int k = kc * 32 + ((l & 16) ? 8 : 0) + ((e & 8) ? 16 : 0) + (e & 7);
    int n = nt * 16 + (l & 15);
    float v = (k < krows && n < ncols) ? W[k * ld + n] : 0.0f;
    dst[d] = f2bf(v);
  }
}

__global__ __launch_bounds__(256, 1) void rln_fused_kernel(
    const float* __restrict__ x,
    const float* __restrict__ wv,  const float* __restrict__ bv,
    const float* __restrict__ wo,  const float* __restrict__ bo,
    const float* __restrict__ wp,  const float* __restrict__ bp,
    const float* __restrict__ ffw, const float* __restrict__ ffb,
    const float* __restrict__ wih, const float* __restrict__ whh,
    const float* __restrict__ lb,
    const float* __restrict__ w1,  const float* __restrict__ b1,
    const float* __restrict__ w2,  const float* __restrict__ b2,
    float* __restrict__ out) {
  extern __shared__ char smem[];
  __bf16* s_seq  = (__bf16*)(smem + SEQ_OFF);
  __bf16* s_whh  = (__bf16*)(smem + WHH_OFF);
  __bf16* s_wih  = (__bf16*)(smem + WIH_OFF);
  __bf16* s_wv   = (__bf16*)(smem + WV_OFF);
  __bf16* s_wo   = (__bf16*)(smem + WO_OFF);
  __bf16* s_wp   = (__bf16*)(smem + WP_OFF);
  __bf16* s_ffw  = (__bf16*)(smem + FFW_OFF);
  float*  s_bias = (float*)(smem + BIAS_OFF);
  float*  s_gate = (float*)(smem + GATE_OFF);   // col-major [256][16]
  __bf16* s_h    = (__bf16*)(smem + HBUF_OFF);  // row-major [16][64]
  __bf16* s_scr  = (__bf16*)(smem + SCR_OFF);
  float*  s_m1   = (float*)(smem + M1_OFF);     // col-major [608][16]
  __bf16* s_w1   = (__bf16*)(smem + W1_OFF);
  float*  s_w2   = (float*)(smem + W2_OFF);

  const int tid   = threadIdx.x;
  const int wave  = tid >> 5;
  const int lane  = tid & 31;
  const int mrow  = lane & 15;
  const int hi    = (lane >> 4) & 1;
  const int base0 = hi ? 8 : 0;   // A-fragment K run base
  const int dm    = hi ? 8 : 0;   // D-fragment M base
  const int bbase = blockIdx.x * 16;

  // ---------- preload: swizzled B weights + biases ----------
  swizzleB(s_wv,  wv,  32,  32, 32,  2,  1, tid);
  swizzleB(s_wo,  wo,  32,  32, 32,  2,  1, tid);
  swizzleB(s_wp,  wp,  16,  32, 16,  1,  1, tid);
  swizzleB(s_ffw, ffw, 256, 32, 256, 16, 1, tid);
  swizzleB(s_wih, wih, 256, 16, 256, 16, 1, tid);  // K 16 -> zero-padded 32
  swizzleB(s_whh, whh, 256, 64, 256, 16, 2, tid);
  if (tid < 32) { s_bias[BV_B + tid] = bv[tid]; s_bias[BO_B + tid] = bo[tid]; }
  if (tid < 16) s_bias[BP_B + tid] = bp[tid];
  if (tid < G4) { s_bias[FFB_B + tid] = ffb[tid]; s_bias[LB_B + tid] = lb[tid]; }
  for (int i = tid; i < DMLP; i += 256) s_bias[B1_B + i] = b1[i];
  if (tid < 8) s_bias[B2_B + tid] = (tid < NCLS) ? b2[tid] : 0.0f;
  for (int i = tid; i < 16 * HL; i += 256) s_h[i] = f2bf(0.0f);  // h0 = 0
  __syncthreads();

  v8bf z8;
#pragma unroll
  for (int e = 0; e < 8; ++e) z8[e] = f2bf(0.0f);

  __bf16* vbuf = s_scr + wave * 1024;  // per-wave [16][32]
  __bf16* abuf = vbuf + 512;           // per-wave [16][32]

  // ---------- Phase A: TFE chain, write seq[:, 0:124] ----------
  for (int c = wave; c < CCH; c += 8) {
    const float* xp = x + ((bbase + mrow) * CCH + c) * FIN;
    v16bf ax;
#pragma unroll
    for (int e = 0; e < 8; ++e) ax[e] = f2bf(xp[base0 + e]);
#pragma unroll
    for (int e = 0; e < 8; ++e) ax[8 + e] = f2bf(xp[base0 + 16 + e]);
    // v = ax @ wv + bv -> vbuf (two independent accumulator chains)
    {
      float b0 = s_bias[BV_B + mrow], b1v = s_bias[BV_B + 16 + mrow];
      v8f acc0, acc1;
#pragma unroll
      for (int r = 0; r < 8; ++r) { acc0[r] = b0; acc1[r] = b1v; }
      v16bf bw0 = *(const v16bf*)(s_wv + lane * 16);
      v16bf bw1 = *(const v16bf*)(s_wv + 512 + lane * 16);
      acc0 = WMMA_BF16(ax, bw0, acc0);
      acc1 = WMMA_BF16(ax, bw1, acc1);
#pragma unroll
      for (int r = 0; r < 8; ++r) {
        vbuf[(r + dm) * 32 + mrow]      = f2bf(acc0[r]);
        vbuf[(r + dm) * 32 + 16 + mrow] = f2bf(acc1[r]);
      }
    }
    asm volatile("s_wait_dscnt 0" ::: "memory");
    v16bf av = combine16(*(const v8bf*)(vbuf + mrow * 32 + base0),
                         *(const v8bf*)(vbuf + mrow * 32 + base0 + 16));
    // attn = av @ wo + bo -> abuf
    {
      float b0 = s_bias[BO_B + mrow], b1v = s_bias[BO_B + 16 + mrow];
      v8f acc0, acc1;
#pragma unroll
      for (int r = 0; r < 8; ++r) { acc0[r] = b0; acc1[r] = b1v; }
      v16bf bw0 = *(const v16bf*)(s_wo + lane * 16);
      v16bf bw1 = *(const v16bf*)(s_wo + 512 + lane * 16);
      acc0 = WMMA_BF16(av, bw0, acc0);
      acc1 = WMMA_BF16(av, bw1, acc1);
#pragma unroll
      for (int r = 0; r < 8; ++r) {
        abuf[(r + dm) * 32 + mrow]      = f2bf(acc0[r]);
        abuf[(r + dm) * 32 + 16 + mrow] = f2bf(acc1[r]);
      }
    }
    asm volatile("s_wait_dscnt 0" ::: "memory");
    v16bf aa = combine16(*(const v8bf*)(abuf + mrow * 32 + base0),
                         *(const v8bf*)(abuf + mrow * 32 + base0 + 16));
    // x1 = aa @ wp + bp -> seq[:, c, :]
    {
      v8f acc;
      float bb = s_bias[BP_B + mrow];
#pragma unroll
      for (int r = 0; r < 8; ++r) acc[r] = bb;
      v16bf bw = *(const v16bf*)(s_wp + lane * 16);
      acc = WMMA_BF16(aa, bw, acc);
#pragma unroll
      for (int r = 0; r < 8; ++r)
        s_seq[((r + dm) * TSEQ + c) * 16 + mrow] = f2bf(acc[r]);
    }
  }

  // ---------- Phase B: FFE, wave w = channel cf; write seq[:, 124:248] ----------
  {
    int cf = wave;  // 0..7
    const float* xp = x + ((bbase + mrow) * CCH + cf) * FIN;
    v16bf axf;
#pragma unroll
    for (int e = 0; e < 8; ++e) axf[e] = f2bf(xp[base0 + e]);
#pragma unroll
    for (int e = 0; e < 8; ++e) axf[8 + e] = f2bf(xp[base0 + 16 + e]);
#pragma unroll
    for (int nt = 0; nt < 16; ++nt) {
      int col = nt * 16 + mrow;
      int tq = 124 + cf * 16 + nt;  // wave-uniform
      v8f acc;
      float bb = s_bias[FFB_B + col];
#pragma unroll
      for (int r = 0; r < 8; ++r) acc[r] = bb;
      v16bf bw = *(const v16bf*)(s_ffw + nt * 512 + lane * 16);
      acc = WMMA_BF16(axf, bw, acc);
      if (tq < TSEQ) {
#pragma unroll
        for (int r = 0; r < 8; ++r)
          s_seq[((r + dm) * TSEQ + tq) * 16 + mrow] = f2bf(leakyf(acc[r]));
      }
    }
  }
  __syncthreads();

  // ---------- Phase C: 248-step LSTM, weights + biases pinned in VGPRs ----------
  v16bf bwih[2], bwhh0[2], bwhh1[2];
  v8f biasr[2];
#pragma unroll
  for (int nn = 0; nn < 2; ++nn) {
    int nt = wave * 2 + nn;
    bwih[nn]  = *(const v16bf*)(s_wih + nt * 512 + lane * 16);
    bwhh0[nn] = *(const v16bf*)(s_whh + (0 * 16 + nt) * 512 + lane * 16);
    bwhh1[nn] = *(const v16bf*)(s_whh + (1 * 16 + nt) * 512 + lane * 16);
    float bb = s_bias[LB_B + nt * 16 + mrow];
#pragma unroll
    for (int r = 0; r < 8; ++r) biasr[nn][r] = bb;
  }
  float creg[4];
#pragma unroll
  for (int q = 0; q < 4; ++q) creg[q] = 0.0f;
  const int urow = (tid * 4) >> 6;        // update-cell row
  const int uj   = (tid * 4) & 63;        // update-cell column base (mult of 4)
  const int nt0  = wave * 2, nt1 = wave * 2 + 1;

  for (int t = 0; t < TSEQ; ++t) {
    // shared A fragments (same for both N-tiles)
    v16bf ax = combine16(*(const v8bf*)(s_seq + (mrow * TSEQ + t) * 16 + base0), z8);
    const __bf16* hrow = s_h + mrow * HL;
    v16bf ah0 = combine16(*(const v8bf*)(hrow + base0),
                          *(const v8bf*)(hrow + base0 + 16));
    v16bf ah1 = combine16(*(const v8bf*)(hrow + 32 + base0),
                          *(const v8bf*)(hrow + 32 + base0 + 16));
    // interleaved independent accumulator chains: no WMMA->WMMA RAW hazard,
    // consecutive instructions share the A operand
    v8f acc0 = biasr[0], acc1 = biasr[1];
    acc0 = WMMA_BF16(ax,  bwih[0],  acc0);
    acc1 = WMMA_BF16(ax,  bwih[1],  acc1);
    acc0 = WMMA_BF16(ah0, bwhh0[0], acc0);
    acc1 = WMMA_BF16(ah0, bwhh0[1], acc1);
    acc0 = WMMA_BF16(ah1, bwhh1[0], acc0);
    acc1 = WMMA_BF16(ah1, bwhh1[1], acc1);
    *(v8f*)(s_gate + (nt0 * 16 + mrow) * 16 + dm) = acc0;  // col-major
    *(v8f*)(s_gate + (nt1 * 16 + mrow) * 16 + dm) = acc1;
    __syncthreads();
    // pointwise cell/hidden update: 4 consecutive cells per thread
    v4bf hv;
#pragma unroll
    for (int q = 0; q < 4; ++q) {
      int j = uj + q;
      float gi = s_gate[(j)       * 16 + urow];
      float gf = s_gate[(64 + j)  * 16 + urow];
      float gg = s_gate[(128 + j) * 16 + urow];
      float go = s_gate[(192 + j) * 16 + urow];
      float cc = sigmf(gf) * creg[q] + sigmf(gi) * tanh_fast(gg);
      creg[q] = cc;
      hv[q] = f2bf(sigmf(go) * tanh_fast(cc));
    }
    *(v4bf*)(s_h + urow * HL + uj) = hv;
    __syncthreads();
  }

  // ---------- Phase D: MLP head (overlay w1/w2 into dead seq region) ----------
  swizzleB(s_w1, w1, DMLP, 64, DMLP, 38, 2, tid);  // col>=600 zero-baked
  for (int i = tid; i < DMLP * NCLS; i += 256) s_w2[i] = w2[i];
  __syncthreads();
  // m1 = leaky(h_last @ w1 + b1): M=16, 38 N-tiles, K=64
  {
    const __bf16* hrow = s_h + mrow * HL;
    v16bf ah0 = combine16(*(const v8bf*)(hrow + base0),
                          *(const v8bf*)(hrow + base0 + 16));
    v16bf ah1 = combine16(*(const v8bf*)(hrow + 32 + base0),
                          *(const v8bf*)(hrow + 32 + base0 + 16));
    for (int nt = wave; nt < 38; nt += 8) {
      int col = nt * 16 + mrow;               // up to 607; OOB cols junk, never read
      v8f acc;
      float bb = s_bias[B1_B + col];          // index <= 1199, in-bounds
#pragma unroll
      for (int r = 0; r < 8; ++r) acc[r] = bb;
      v16bf bw0 = *(const v16bf*)(s_w1 + (0 * 38 + nt) * 512 + lane * 16);
      v16bf bw1 = *(const v16bf*)(s_w1 + (1 * 38 + nt) * 512 + lane * 16);
      acc = WMMA_BF16(ah0, bw0, acc);
      acc = WMMA_BF16(ah1, bw1, acc);
      v8f res;
#pragma unroll
      for (int r = 0; r < 8; ++r) res[r] = leakyf(acc[r]);
      *(v8f*)(s_m1 + col * 16 + dm) = res;    // col-major [608][16]
    }
  }
  __syncthreads();
  // out = leaky(m1 @ w2 + b2): 16x6 outputs, scalar dot per thread
  if (tid < 16 * NCLS) {
    int r = tid / NCLS, cls = tid % NCLS;
    float s = s_bias[B2_B + cls];
    for (int k = 0; k < DMLP; ++k) s += s_m1[k * 16 + r] * s_w2[k * NCLS + cls];
    out[(bbase + r) * NCLS + cls] = leakyf(s);
  }
}

extern "C" void kernel_launch(void* const* d_in, const int* in_sizes, int n_in,
                              void* d_out, int out_size, void* d_ws, size_t ws_size,
                              hipStream_t stream) {
  (void)in_sizes; (void)n_in; (void)out_size; (void)d_ws; (void)ws_size;
  const float* x   = (const float*)d_in[0];
  const float* wv  = (const float*)d_in[1];
  const float* bv  = (const float*)d_in[2];
  const float* wo  = (const float*)d_in[3];
  const float* bo  = (const float*)d_in[4];
  const float* wp  = (const float*)d_in[5];
  const float* bp  = (const float*)d_in[6];
  const float* ffw = (const float*)d_in[7];
  const float* ffb = (const float*)d_in[8];
  const float* wih = (const float*)d_in[9];
  const float* whh = (const float*)d_in[10];
  const float* lb  = (const float*)d_in[11];
  const float* w1  = (const float*)d_in[12];
  const float* b1  = (const float*)d_in[13];
  const float* w2  = (const float*)d_in[14];
  const float* b2  = (const float*)d_in[15];

  (void)hipFuncSetAttribute((const void*)rln_fused_kernel,
                            hipFuncAttributeMaxDynamicSharedMemorySize, LDS_BYTES);
  dim3 grid(BSZ / 16), block(256);
  rln_fused_kernel<<<grid, block, LDS_BYTES, stream>>>(
      x, wv, bv, wo, bo, wp, bp, ffw, ffb, wih, whh, lb, w1, b1, w2, b2,
      (float*)d_out);
}
